// ReductionBlock_16123307229656
// MI455X (gfx1250) — 8-bit matrix operand order repaired and checked by reading, compiled, not run on hardware
//
#include <hip/hip_runtime.h>
#include <hip/hip_bf16.h>

typedef int   v8i __attribute__((ext_vector_type(8)));
typedef int   v4i __attribute__((ext_vector_type(4)));
typedef int   v2i __attribute__((ext_vector_type(2)));
typedef float v4f __attribute__((ext_vector_type(4)));
typedef signed char s8;

#define B_    32
#define C_    256
#define H_    56
#define W_    56
#define HP_   58
#define HWP_  (HP_*HP_)
#define OW_   28
#define HW_   (H_*W_)
#define PHW   (28*28)
#define NPIX  (B_*PHW)
#define CO1_  512
#define EPSF  1e-5f

#if defined(__has_builtin)
# if __has_builtin(__builtin_amdgcn_global_load_async_to_lds_b128) && \
     __has_builtin(__builtin_amdgcn_s_wait_asynccnt)
#  define USE_ASYNC_LDS 1
# endif
#endif

typedef __attribute__((address_space(1))) v4i gas_v4i;
typedef __attribute__((address_space(3))) v4i las_v4i;

__global__ void k_binarize_x(const float* __restrict__ x, s8* __restrict__ xb) {
  __shared__ __align__(16) s8 lds[64 * 64];
  int t  = threadIdx.x;
  int b  = blockIdx.z;
  int c0 = blockIdx.y * 64;
  int p0 = blockIdx.x * 64;
  const float* src = x + ((size_t)b * C_ + c0) * HW_ + p0;
  #pragma unroll
  for (int i = 0; i < 16; ++i) {
    int idx = t + i * 256;
    int cl = idx >> 6, pl = idx & 63;
    float v = src[(size_t)cl * HW_ + pl];
    lds[pl * 64 + cl] = (v >= 0.0f) ? (s8)1 : (s8)-1;
  }
  __syncthreads();
  int pl = t >> 2, q = t & 3;
  int p  = p0 + pl;
  int hh = p / W_, ww = p % W_;
  v4i d = *(const v4i*)(lds + pl * 64 + q * 16);
  *(v4i*)(xb + (size_t)(b * HWP_ + (hh + 1) * HP_ + (ww + 1)) * C_ + c0 + q * 16) = d;
}

__global__ void k_zero_border(s8* __restrict__ xb) {
  int id = blockIdx.x * 256 + threadIdx.x;
  int q  = id & 15;
  int r  = id >> 4;
  int b  = r / 228, bp = r % 228;
  int hh, ww;
  if (bp < 58)       { hh = 0;            ww = bp; }
  else if (bp < 116) { hh = 57;           ww = bp - 58; }
  else if (bp < 172) { hh = bp - 116 + 1; ww = 0; }
  else               { hh = bp - 172 + 1; ww = 57; }
  *(v4i*)(xb + (size_t)(b * HWP_ + hh * HP_ + ww) * C_ + q * 16) = (v4i){};
}

__global__ void k_avgpool(const float* __restrict__ x, float* __restrict__ pool) {
  int id = blockIdx.x * 256 + threadIdx.x;
  int c = id & 255;
  int n = id >> 8;
  int b = n / PHW, p = n % PHW;
  int oh = p / OW_, ow = p % OW_;
  const float* src = x + (((size_t)b * C_ + c) * H_ + oh * 2) * W_ + ow * 2;
  pool[(size_t)n * C_ + c] = 0.25f * (src[0] + src[1] + src[W_] + src[W_ + 1]);
}

__global__ void k_pack_w3(const float* __restrict__ w3, s8* __restrict__ wp3) {
  int id = blockIdx.x * 256 + threadIdx.x;
  int bpos = id & 3;
  int v    = (id >> 2) & 7;
  int l    = (id >> 5) & 31;
  int kc   = (id >> 10) % 36;
  int ct   = id / (36 * 1024);
  int klocal = ((l >> 4) * 8) + ((v & 1) * 4) + bpos + ((v >> 1) * 16);
  int co  = ct * 16 + (l & 15);
  int khw = kc >> 2;
  int ci  = (kc & 3) * 64 + klocal;
  float w = w3[((size_t)co * C_ + ci) * 9 + khw];
  wp3[id] = (w >= 0.0f) ? (s8)1 : (s8)-1;
}

__global__ void k_pack_w1(const float* __restrict__ w1, s8* __restrict__ wp1) {
  int id = blockIdx.x * 256 + threadIdx.x;
  int bpos = id & 3;
  int v    = (id >> 2) & 7;
  int l    = (id >> 5) & 31;
  int kc   = (id >> 10) & 3;
  int ct   = id >> 12;
  int k  = ((l >> 4) * 16) + ((v & 3) * 4) + ((v >> 2) * 32) + bpos;
  int co = ct * 16 + (l & 15);
  int ci = kc * 64 + k;
  float w = w1[(size_t)co * C_ + ci];
  wp1[id] = (w >= 0.0f) ? (s8)1 : (s8)-1;
}

__global__ void __launch_bounds__(256, 2)
k_conv3(const s8* __restrict__ xb, const s8* __restrict__ wp3,
        const float* __restrict__ pool,
        const float* __restrict__ g3, const float* __restrict__ b3,
        const float* __restrict__ m3, const float* __restrict__ v3,
        float* __restrict__ h, s8* __restrict__ hb) {
  __shared__ __align__(16) s8 wlds[73728];
  int t = threadIdx.x;

  {
    const s8* wsrc = wp3 + (size_t)blockIdx.y * 73728;
    #pragma unroll
    for (int i = 0; i < 18; ++i) {
      int off = t * 16 + i * 4096;
#if USE_ASYNC_LDS
      __builtin_amdgcn_global_load_async_to_lds_b128(
          (gas_v4i*)(wsrc + off), (las_v4i*)(wlds + off), 0, 0);
#else
      *(v4i*)(wlds + off) = *(const v4i*)(wsrc + off);
#endif
    }
#if USE_ASYNC_LDS
    __builtin_amdgcn_s_wait_asynccnt(0);
#endif
  }
  __syncthreads();

  int lane = t & 31;
  int wave = t >> 5;
  int px0 = (blockIdx.x * 8 + wave) * 16;
  int co0 = blockIdx.y * 32;

  int apx = px0 + (lane & 15);
  int ab  = apx / PHW;
  int ap  = apx % PHW;
  int aoh = ap / OW_, aow = ap % OW_;
  const s8* abase = xb + (size_t)(ab * HWP_ + aoh * 2 * HP_ + aow * 2) * C_ + ((lane >> 4) * 8);
  const s8* wl0 = wlds + lane * 32;
  const s8* wl1 = wlds + 36864 + lane * 32;

  v8i acc0 = {}; v8i acc1 = {};

  #pragma unroll
  for (int kh = 0; kh < 3; ++kh) {
    #pragma unroll
    for (int kw = 0; kw < 3; ++kw) {
      #pragma unroll
      for (int j = 0; j < 4; ++j) {
        const int off = (kh * HP_ + kw) * C_ + j * 64;
        const v2i* ap8 = (const v2i*)(abase + off);
        v2i d0 = ap8[0];
        v2i d1 = ap8[2];
        v2i d2 = ap8[4];
        v2i d3 = ap8[6];
        v8i a;
        a[0] = d0[0]; a[1] = d0[1]; a[2] = d1[0]; a[3] = d1[1];
        a[4] = d2[0]; a[5] = d2[1]; a[6] = d3[0]; a[7] = d3[1];
        const int kcb = ((kh * 3 + kw) * 4 + j) * 1024;
        v8i bf0 = *(const v8i*)(wl0 + kcb);
        v8i bf1 = *(const v8i*)(wl1 + kcb);
        acc0 = __builtin_amdgcn_wmma_i32_16x16x64_iu8(true, a, true, bf0, acc0, false, false);
        acc1 = __builtin_amdgcn_wmma_i32_16x16x64_iu8(true, a, true, bf1, acc1, false, false);
      }
    }
  }

  #pragma unroll
  for (int s = 0; s < 2; ++s) {
    v8i acc = s ? acc1 : acc0;
    int co = co0 + s * 16 + (lane & 15);
    float inv  = g3[co] * rsqrtf(v3[co] + EPSF);
    float bias = b3[co] - m3[co] * inv;
    #pragma unroll
    for (int r = 0; r < 8; ++r) {
      int px = px0 + r + ((lane >> 4) * 8);
      size_t idx = (size_t)px * C_ + co;
      float hv = (float)acc[r] * inv + bias + pool[idx];
      h[idx]  = hv;
      hb[idx] = (hv >= 0.0f) ? (s8)1 : (s8)-1;
    }
  }
}

__global__ void __launch_bounds__(256, 2)
k_conv1(const s8* __restrict__ hb, const s8* __restrict__ wp1,
        const float* __restrict__ h,
        const float* __restrict__ g1, const float* __restrict__ b1,
        const float* __restrict__ m1, const float* __restrict__ v1,
        float* __restrict__ out) {
  __shared__ __align__(16) s8 wlds[4096];
  int t = threadIdx.x;
  {
    const s8* wsrc = wp1 + (size_t)blockIdx.y * 4096;
#if USE_ASYNC_LDS
    __builtin_amdgcn_global_load_async_to_lds_b128(
        (gas_v4i*)(wsrc + t * 16), (las_v4i*)(wlds + t * 16), 0, 0);
    __builtin_amdgcn_s_wait_asynccnt(0);
#else
    *(v4i*)(wlds + t * 16) = *(const v4i*)(wsrc + t * 16);
#endif
  }
  __syncthreads();

  int lane = t & 31;
  int wave = t >> 5;
  int px0 = (blockIdx.x * 8 + wave) * 16;
  int co0 = blockIdx.y * 16;

  int bpx = px0 + (lane & 15);
  const s8* bbase = hb + (size_t)bpx * C_ + ((lane >> 4) * 16);
  const s8* abase = wlds + lane * 32;

  v8i acc = {};
  #pragma unroll
  for (int kc = 0; kc < 4; ++kc) {
    v8i a = *(const v8i*)(abase + kc * 1024);
    const v4i* bp = (const v4i*)(bbase + kc * 64);
    v4i blo = bp[0];
    v4i bhi = bp[2];
    v8i bf;
    bf[0] = blo[0]; bf[1] = blo[1]; bf[2] = blo[2]; bf[3] = blo[3];
    bf[4] = bhi[0]; bf[5] = bhi[1]; bf[6] = bhi[2]; bf[7] = bhi[3];
    acc = __builtin_amdgcn_wmma_i32_16x16x64_iu8(true, a, true, bf, acc, false, false);
  }

  int pb = bpx / PHW, pp = bpx % PHW;
  int rbase = co0 + ((lane >> 4) * 8);
  v4f gl = *(const v4f*)(g1 + rbase), gh = *(const v4f*)(g1 + rbase + 4);
  v4f bl = *(const v4f*)(b1 + rbase), bh = *(const v4f*)(b1 + rbase + 4);
  v4f ml = *(const v4f*)(m1 + rbase), mh = *(const v4f*)(m1 + rbase + 4);
  v4f vl = *(const v4f*)(v1 + rbase), vh = *(const v4f*)(v1 + rbase + 4);
  size_t hoff = (size_t)bpx * C_ + (rbase & 255);
  v4f hl = *(const v4f*)(h + hoff), hh2 = *(const v4f*)(h + hoff + 4);
  float* obase = out + ((size_t)pb * CO1_ + rbase) * PHW + pp;
  #pragma unroll
  for (int r = 0; r < 8; ++r) {
    float g = (r < 4) ? gl[r & 3] : gh[r & 3];
    float bb = (r < 4) ? bl[r & 3] : bh[r & 3];
    float mm = (r < 4) ? ml[r & 3] : mh[r & 3];
    float vv = (r < 4) ? vl[r & 3] : vh[r & 3];
    float hv = (r < 4) ? hl[r & 3] : hh2[r & 3];
    float inv = g * rsqrtf(vv + EPSF);
    obase[(size_t)r * PHW] = (float)acc[r] * inv + (bb - mm * inv) + hv;
  }
}

extern "C" void kernel_launch(void* const* d_in, const int* in_sizes, int n_in,
                              void* d_out, int out_size, void* d_ws, size_t ws_size,
                              hipStream_t stream) {
  const float* x  = (const float*)d_in[0];
  const float* w3 = (const float*)d_in[1];
  const float* g3 = (const float*)d_in[2];
  const float* b3 = (const float*)d_in[3];
  const float* m3 = (const float*)d_in[4];
  const float* v3 = (const float*)d_in[5];
  const float* w1 = (const float*)d_in[6];
  const float* g1 = (const float*)d_in[7];
  const float* b1 = (const float*)d_in[8];
  const float* m1 = (const float*)d_in[9];
  const float* v1 = (const float*)d_in[10];

  char* ws = (char*)d_ws;
  s8*    xb   = (s8*)ws;
  float* pool = (float*)(ws + 27557888);
  float* h    = (float*)(ws + 53248000);
  s8*    hbuf = (s8*)(ws + 78938112);
  s8*    wp3  = (s8*)(ws + 85360640);
  s8*    wp1  = (s8*)(ws + 85950464);

  k_zero_border<<<456, 256, 0, stream>>>(xb);
  k_binarize_x<<<dim3(49, 4, 32), 256, 0, stream>>>(x, xb);
  k_avgpool<<<(NPIX * C_) / 256, 256, 0, stream>>>(x, pool);
  k_pack_w3<<<589824 / 256, 256, 0, stream>>>(w3, wp3);
  k_pack_w1<<<131072 / 256, 256, 0, stream>>>(w1, wp1);
  k_conv3<<<dim3(196, 8), 256, 0, stream>>>(xb, wp3, pool, g3, b3, m3, v3, h, hbuf);
  k_conv1<<<dim3(196, 32), 256, 0, stream>>>(hbuf, wp1, h, g1, b1, m1, v1, (float*)d_out);
}
